// MultiHeadedAttentionWithAdapter_31078383354325
// MI455X (gfx1250) — matrix operand order repaired and checked by reading, not compiled, not run on hardware
//
#include <hip/hip_runtime.h>
#include <math.h>

// ---------------------------------------------------------------------------
// Problem constants (from reference): B=8, S=1024, D=1024, H=16, HD=64
// ---------------------------------------------------------------------------
#define B_NUM 8
#define S_DIM 1024
#define D_DIM 1024
#define H_NUM 16
#define HD_DIM 64

typedef __bf16  v16bf  __attribute__((ext_vector_type(16)));
typedef __bf16  bf16x8 __attribute__((ext_vector_type(8)));
typedef float   v8f    __attribute__((ext_vector_type(8)));
typedef float   f32x4  __attribute__((ext_vector_type(4)));

static __device__ __forceinline__ v8f wmma_bf16(v16bf a, v16bf b, v8f c) {
  // D = A(16x32 bf16) * B(32x16 bf16) + C(16x16 f32)
  return __builtin_amdgcn_wmma_f32_16x16x32_bf16(
      /*neg_a=*/false, a, /*neg_b=*/false, b,
      /*c_mod=*/(short)0, c, /*reuse_a=*/false, /*reuse_b=*/false);
}

// B-operand fragment: lane (l&15) = column, lanes<16 hold K {0-7,16-23}, lanes>=16 hold K {8-15,24-31}.
// Source must be K-contiguous at `col_ptr` (the column's 32 K values).
static __device__ __forceinline__ v16bf load_bfrag(const __bf16* col_ptr, int kh16) {
  bf16x8 b0 = *(const bf16x8*)(col_ptr + (kh16 >> 1));
  bf16x8 b1 = *(const bf16x8*)(col_ptr + 16 + (kh16 >> 1));
  v16bf r;
#pragma unroll
  for (int i = 0; i < 8; ++i) { r[i] = b0[i]; r[i + 8] = b1[i]; }
  return r;
}

// A-operand fragment from bf16 K-contiguous row: lane (l&15) = row,
// lanes<16 hold K {0-7,16-23}, lanes>=16 hold K {8-15,24-31}.
static __device__ __forceinline__ v16bf load_afrag_bf16(const __bf16* row_ptr, int kh8) {
  bf16x8 a0 = *(const bf16x8*)(row_ptr + kh8);
  bf16x8 a1 = *(const bf16x8*)(row_ptr + 16 + kh8);
  v16bf r;
#pragma unroll
  for (int i = 0; i < 8; ++i) { r[i] = a0[i]; r[i + 8] = a1[i]; }
  return r;
}

// A-operand fragment from fp32 row (convert on the fly).
static __device__ __forceinline__ v16bf load_afrag_f32(const float* row_ptr, int kh8) {
  f32x4 a0 = *(const f32x4*)(row_ptr + kh8);
  f32x4 a1 = *(const f32x4*)(row_ptr + kh8 + 4);
  f32x4 a2 = *(const f32x4*)(row_ptr + 16 + kh8);
  f32x4 a3 = *(const f32x4*)(row_ptr + 16 + kh8 + 4);
  v16bf r;
#pragma unroll
  for (int i = 0; i < 4; ++i) {
    r[i]      = (__bf16)a0[i];
    r[4 + i]  = (__bf16)a1[i];
    r[8 + i]  = (__bf16)a2[i];
    r[12 + i] = (__bf16)a3[i];
  }
  return r;
}

// ---------------------------------------------------------------------------
// Kernel 0: fp32 -> bf16 weight conversion
// ---------------------------------------------------------------------------
__global__ __launch_bounds__(256)
void cvt_f32_to_bf16(const float* __restrict__ src, __bf16* __restrict__ dst, int n) {
  int i = (blockIdx.x * 256 + threadIdx.x) * 4;
  if (i + 3 < n) {
    f32x4 v = *(const f32x4*)(src + i);
#pragma unroll
    for (int j = 0; j < 4; ++j) dst[i + j] = (__bf16)v[j];
  }
}

// ---------------------------------------------------------------------------
// Kernel 1: input projection  C[m,n] = sum_k A[m,k]*W[n,k] + bias[n]
//   A: (B*S, D) fp32 (converted to bf16 frags on the fly)
//   W: (D, D) bf16 (rows of W are columns of the B operand -> K-contiguous)
//   tmode 0/1: write (B,H,S,HD); tmode 2: write V transposed (B,H,HD,S)
// Block = 256 threads (8 waves). Tile: M=128 (16 per wave), N=64 (4 subtiles).
// K-loop is 2x ping-pong unrolled: fragment set B is prefetched while set A
// feeds the WMMAs (and vice versa) -- no buffer-rotation register copies.
// ---------------------------------------------------------------------------
__global__ __launch_bounds__(256)
void proj_gemm(const float* __restrict__ A, const __bf16* __restrict__ W,
               const float* __restrict__ bias, __bf16* __restrict__ out, int tmode) {
  const int tid  = threadIdx.x;
  const int wave = tid >> 5;
  const int lane = tid & 31;
  const int l15  = lane & 15;
  const int kh8  = (lane >> 4) << 3;   // 0 or 8
  const int kh16 = (lane >> 4) << 4;   // 0 or 16
  const int Mbase = blockIdx.x * 128 + wave * 16;
  const int Nbase = blockIdx.y * 64;

  const float*  arow  = A + (size_t)(Mbase + l15) * D_DIM;
  const __bf16* wrow0 = W + (size_t)(Nbase + l15) * D_DIM;

  v8f zero = {};
  v8f acc[4];
#pragma unroll
  for (int f = 0; f < 4; ++f) acc[f] = zero;

  // Prologue: fragment set A for kb = 0
  v16bf afA = load_afrag_f32(arow, kh8);
  v16bf bfA[4];
#pragma unroll
  for (int sub = 0; sub < 4; ++sub)
    bfA[sub] = load_bfrag(wrow0 + (size_t)sub * 16 * D_DIM, kh16);

  for (int kb = 0; kb < D_DIM; kb += 64) {
    // Prefetch set B (kb+32) behind set A's WMMAs.
    v16bf afB = load_afrag_f32(arow + kb + 32, kh8);
    v16bf bfB[4];
#pragma unroll
    for (int sub = 0; sub < 4; ++sub)
      bfB[sub] = load_bfrag(wrow0 + (size_t)sub * 16 * D_DIM + kb + 32, kh16);

#pragma unroll
    for (int sub = 0; sub < 4; ++sub)
      acc[sub] = wmma_bf16(afA, bfA[sub], acc[sub]);

    // Prefetch set A (kb+64; clamped to 0 on last step) behind set B's WMMAs.
    const int kn = (kb + 64 < D_DIM) ? (kb + 64) : 0;
    afA = load_afrag_f32(arow + kn, kh8);
#pragma unroll
    for (int sub = 0; sub < 4; ++sub)
      bfA[sub] = load_bfrag(wrow0 + (size_t)sub * 16 * D_DIM + kn, kh16);

#pragma unroll
    for (int sub = 0; sub < 4; ++sub)
      acc[sub] = wmma_bf16(afB, bfB[sub], acc[sub]);
  }

#pragma unroll
  for (int sub = 0; sub < 4; ++sub) {
    const int n  = Nbase + sub * 16 + l15;
    const float bv = bias[n];
    const int hh = n >> 6, dd = n & 63;
#pragma unroll
    for (int r = 0; r < 8; ++r) {
      const int m    = Mbase + r + ((lane >> 4) << 3);
      const int bidx = m >> 10;
      const int srow = m & 1023;
      const float v  = acc[sub][r] + bv;
      size_t addr;
      if (tmode == 2)   // V transposed: (B,H,HD,S)
        addr = ((size_t)(bidx * H_NUM + hh) << 16) + ((size_t)dd << 10) + srow;
      else              // Q/K: (B,H,S,HD)
        addr = ((size_t)(bidx * H_NUM + hh) << 16) + ((size_t)srow << 6) + dd;
      out[addr] = (__bf16)v;
    }
  }
}

// ---------------------------------------------------------------------------
// Kernel 2: fused attention (flash-style, online softmax)
// grid = (B*H, S/128); block = 256 = 8 waves; wave owns 16 query rows.
// K fragments are double-buffered across key blocks; syn/mask gathers are
// issued before the score WMMAs so every memory op overlaps matrix/VALU work.
// ---------------------------------------------------------------------------
__global__ __launch_bounds__(256)
void attn_kernel(const __bf16* __restrict__ Qh, const __bf16* __restrict__ Kh,
                 const __bf16* __restrict__ Vt, const float* __restrict__ syn,
                 const int* __restrict__ mask, const float* __restrict__ alpha_p,
                 __bf16* __restrict__ Abuf) {
  __shared__ __bf16 pbuf[8][16][32];   // per-wave P staging (C-layout -> A-layout)

  const int tid  = threadIdx.x;
  const int wave = tid >> 5;
  const int lane = tid & 31;
  const int l15  = lane & 15;
  const int kh8  = (lane >> 4) << 3;
  const int kh16 = (lane >> 4) << 4;
  const int hh   = blockIdx.x & 15;
  const int bb   = blockIdx.x >> 4;
  const int qbase = blockIdx.y * 128 + wave * 16;
  const size_t head_off = (size_t)blockIdx.x << 16;   // bh * S*HD

  const __bf16* Q = Qh + head_off;
  const __bf16* K = Kh + head_off;
  const __bf16* V = Vt + head_off;

  const float ap       = alpha_p[0];
  const float alpha    = 1.0f / (1.0f + __expf(-ap));
  const float sscale   = alpha * 0.125f;     // alpha / sqrt(HD=64)
  const float synscale = 1.0f - alpha;

  // Q fragments (held for the whole key loop): K-dim = HD = 64 -> 2 chunks
  v16bf qa[2];
  {
    const __bf16* qr = Q + (size_t)(qbase + l15) * HD_DIM;
#pragma unroll
    for (int c = 0; c < 2; ++c) qa[c] = load_afrag_bf16(qr + c * 32, kh8);
  }

  float mrow[8], lrow[8];
  v8f zero = {};
  v8f oacc[4];
#pragma unroll
  for (int r = 0; r < 8; ++r) { mrow[r] = -3.0e38f; lrow[r] = 0.0f; }
#pragma unroll
  for (int f = 0; f < 4; ++f) oacc[f] = zero;

  // Prefetch K fragments for key block 0: kf[sub][c]
  v16bf kf[2][2];
#pragma unroll
  for (int sub = 0; sub < 2; ++sub) {
    const __bf16* kr = K + (size_t)(sub * 16 + l15) * HD_DIM;
#pragma unroll
    for (int c = 0; c < 2; ++c) kf[sub][c] = load_bfrag(kr + c * 32, kh16);
  }

  for (int kb = 0; kb < S_DIM; kb += 32) {
    const int kn = (kb + 32 < S_DIM) ? (kb + 32) : kb;

    // ---- issue syn / mask gathers for this block (independent of WMMA) ----
    float sy[2][8];
    int   mk[2][8];
#pragma unroll
    for (int sub = 0; sub < 2; ++sub) {
      const int key = kb + sub * 16 + l15;
#pragma unroll
      for (int r = 0; r < 8; ++r) {
        const int qrow = qbase + r + ((lane >> 4) << 3);
        sy[sub][r] = syn[((size_t)hh << 20) + ((size_t)qrow << 10) + key];
        mk[sub][r] = mask[((size_t)bb << 20) + ((size_t)qrow << 10) + key];
      }
    }

    // ---- scores = Q * K^T using prefetched K fragments ----
    v8f sacc[2];
    sacc[0] = zero; sacc[1] = zero;
#pragma unroll
    for (int sub = 0; sub < 2; ++sub)
#pragma unroll
      for (int c = 0; c < 2; ++c)
        sacc[sub] = wmma_bf16(qa[c], kf[sub][c], sacc[sub]);

    // ---- prefetch next block's K fragments (overlaps softmax below) ----
    v16bf kf_n[2][2];
#pragma unroll
    for (int sub = 0; sub < 2; ++sub) {
      const __bf16* kr = K + (size_t)(kn + sub * 16 + l15) * HD_DIM;
#pragma unroll
      for (int c = 0; c < 2; ++c) kf_n[sub][c] = load_bfrag(kr + c * 32, kh16);
    }

    // ---- blend with synthesizer scores + mask ----
    float sv[2][8];
#pragma unroll
    for (int sub = 0; sub < 2; ++sub)
#pragma unroll
      for (int r = 0; r < 8; ++r) {
        float s = sacc[sub][r] * sscale + synscale * sy[sub][r];
        sv[sub][r] = (mk[sub][r] == 0) ? -1.0e9f : s;
      }

    // ---- online softmax update (row reductions within 16-lane halves) ----
#pragma unroll
    for (int r = 0; r < 8; ++r) {
      float v = fmaxf(sv[0][r], sv[1][r]);
#pragma unroll
      for (int off = 8; off >= 1; off >>= 1) v = fmaxf(v, __shfl_xor(v, off, 32));
      const float mnew  = fmaxf(mrow[r], v);
      const float scale = __expf(mrow[r] - mnew);
      mrow[r] = mnew;
      lrow[r] *= scale;
#pragma unroll
      for (int f = 0; f < 4; ++f) oacc[f][r] *= scale;
      const float p0 = __expf(sv[0][r] - mnew);
      const float p1 = __expf(sv[1][r] - mnew);
      sv[0][r] = p0; sv[1][r] = p1;
      float rs = p0 + p1;
#pragma unroll
      for (int off = 8; off >= 1; off >>= 1) rs += __shfl_xor(rs, off, 32);
      lrow[r] += rs;
    }

    // ---- P: C-layout -> A-layout via per-wave LDS ----
#pragma unroll
    for (int sub = 0; sub < 2; ++sub)
#pragma unroll
      for (int r = 0; r < 8; ++r) {
        const int row = r + ((lane >> 4) << 3);
        pbuf[wave][row][sub * 16 + l15] = (__bf16)sv[sub][r];
      }

    // ---- V fragments (issued while LDS stores retire) ----
    v16bf vf[4];
#pragma unroll
    for (int f = 0; f < 4; ++f)
      vf[f] = load_bfrag(V + ((size_t)(f * 16 + l15) << 10) + kb, kh16);

    asm volatile("s_wait_dscnt 0" ::: "memory");
    v16bf pa = load_afrag_bf16(&pbuf[wave][l15][0], kh8);

    // ---- O += P * V ----
#pragma unroll
    for (int f = 0; f < 4; ++f)
      oacc[f] = wmma_bf16(pa, vf[f], oacc[f]);

#pragma unroll
    for (int sub = 0; sub < 2; ++sub)
#pragma unroll
      for (int c = 0; c < 2; ++c) kf[sub][c] = kf_n[sub][c];
  }

  // ---- normalize and store to (B*S, D) bf16 buffer ----
#pragma unroll
  for (int r = 0; r < 8; ++r) {
    const float inv = 1.0f / lrow[r];
    const int qrow  = qbase + r + ((lane >> 4) << 3);
    const size_t rowbase = (((size_t)bb << 10) + qrow) * D_DIM + (size_t)hh * 64;
#pragma unroll
    for (int f = 0; f < 4; ++f)
      Abuf[rowbase + f * 16 + l15] = (__bf16)(oacc[f][r] * inv);
  }
}

// ---------------------------------------------------------------------------
// Kernel 3: output projection  out[m,n] = sum_k Ab[m,k]*Wo[n,k] + bo[n]  (fp32 out)
// Same 2x ping-pong K-loop as proj_gemm.
// ---------------------------------------------------------------------------
__global__ __launch_bounds__(256)
void out_gemm(const __bf16* __restrict__ A, const __bf16* __restrict__ W,
              const float* __restrict__ bias, float* __restrict__ out) {
  const int tid  = threadIdx.x;
  const int wave = tid >> 5;
  const int lane = tid & 31;
  const int l15  = lane & 15;
  const int kh8  = (lane >> 4) << 3;
  const int kh16 = (lane >> 4) << 4;
  const int Mbase = blockIdx.x * 128 + wave * 16;
  const int Nbase = blockIdx.y * 64;

  const __bf16* arow  = A + (size_t)(Mbase + l15) * D_DIM;
  const __bf16* wrow0 = W + (size_t)(Nbase + l15) * D_DIM;

  v8f zero = {};
  v8f acc[4];
#pragma unroll
  for (int f = 0; f < 4; ++f) acc[f] = zero;

  v16bf afA = load_afrag_bf16(arow, kh8);
  v16bf bfA[4];
#pragma unroll
  for (int sub = 0; sub < 4; ++sub)
    bfA[sub] = load_bfrag(wrow0 + (size_t)sub * 16 * D_DIM, kh16);

  for (int kb = 0; kb < D_DIM; kb += 64) {
    v16bf afB = load_afrag_bf16(arow + kb + 32, kh8);
    v16bf bfB[4];
#pragma unroll
    for (int sub = 0; sub < 4; ++sub)
      bfB[sub] = load_bfrag(wrow0 + (size_t)sub * 16 * D_DIM + kb + 32, kh16);

#pragma unroll
    for (int sub = 0; sub < 4; ++sub)
      acc[sub] = wmma_bf16(afA, bfA[sub], acc[sub]);

    const int kn = (kb + 64 < D_DIM) ? (kb + 64) : 0;
    afA = load_afrag_bf16(arow + kn, kh8);
#pragma unroll
    for (int sub = 0; sub < 4; ++sub)
      bfA[sub] = load_bfrag(wrow0 + (size_t)sub * 16 * D_DIM + kn, kh16);

#pragma unroll
    for (int sub = 0; sub < 4; ++sub)
      acc[sub] = wmma_bf16(afB, bfB[sub], acc[sub]);
  }

#pragma unroll
  for (int sub = 0; sub < 4; ++sub) {
    const int n    = Nbase + sub * 16 + l15;
    const float bv = bias[n];
#pragma unroll
    for (int r = 0; r < 8; ++r) {
      const int m = Mbase + r + ((lane >> 4) << 3);
      out[(size_t)m * D_DIM + n] = acc[sub][r] + bv;
    }
  }
}

// ---------------------------------------------------------------------------
// Launch
// ---------------------------------------------------------------------------
extern "C" void kernel_launch(void* const* d_in, const int* in_sizes, int n_in,
                              void* d_out, int out_size, void* d_ws, size_t ws_size,
                              hipStream_t stream) {
  (void)in_sizes; (void)n_in; (void)out_size; (void)ws_size;

  const float* query   = (const float*)d_in[0];
  const float* key_in  = (const float*)d_in[1];
  const float* value   = (const float*)d_in[2];
  const int*   mask    = (const int*)d_in[3];
  const float* Wq      = (const float*)d_in[4];
  const float* bq      = (const float*)d_in[5];
  const float* Wk      = (const float*)d_in[6];
  const float* bk      = (const float*)d_in[7];
  const float* Wv      = (const float*)d_in[8];
  const float* bv      = (const float*)d_in[9];
  const float* Wo      = (const float*)d_in[10];
  const float* bo      = (const float*)d_in[11];
  const float* syn     = (const float*)d_in[12];
  const float* alpha_p = (const float*)d_in[13];
  float* out = (float*)d_out;

  char* ws = (char*)d_ws;
  __bf16* wq_bf = (__bf16*)(ws + ((size_t)0  << 20));
  __bf16* wk_bf = (__bf16*)(ws + ((size_t)2  << 20));
  __bf16* wv_bf = (__bf16*)(ws + ((size_t)4  << 20));
  __bf16* wo_bf = (__bf16*)(ws + ((size_t)6  << 20));
  __bf16* Qh    = (__bf16*)(ws + ((size_t)8  << 20));   // (B,H,S,HD)  16 MB
  __bf16* Kh    = (__bf16*)(ws + ((size_t)24 << 20));   // (B,H,S,HD)  16 MB
  __bf16* Vt    = (__bf16*)(ws + ((size_t)40 << 20));   // (B,H,HD,S)  16 MB
  __bf16* Ab    = (__bf16*)(ws + ((size_t)56 << 20));   // (B*S, D)    16 MB

  const int nW = 1 << 20;
  cvt_f32_to_bf16<<<1024, 256, 0, stream>>>(Wq, wq_bf, nW);
  cvt_f32_to_bf16<<<1024, 256, 0, stream>>>(Wk, wk_bf, nW);
  cvt_f32_to_bf16<<<1024, 256, 0, stream>>>(Wv, wv_bf, nW);
  cvt_f32_to_bf16<<<1024, 256, 0, stream>>>(Wo, wo_bf, nW);

  dim3 gg(64, 16);   // M tiles (8192/128) x N tiles (1024/64)
  proj_gemm<<<gg, 256, 0, stream>>>(query,  wq_bf, bq, Qh, 0);
  proj_gemm<<<gg, 256, 0, stream>>>(key_in, wk_bf, bk, Kh, 1);
  proj_gemm<<<gg, 256, 0, stream>>>(value,  wv_bf, bv, Vt, 2);

  attn_kernel<<<dim3(B_NUM * H_NUM, S_DIM / 128), 256, 0, stream>>>(
      Qh, Kh, Vt, syn, mask, alpha_p, Ab);

  out_gemm<<<gg, 256, 0, stream>>>(Ab, wo_bf, bo, out);
}
